// GnBlock_1477468750141
// MI455X (gfx1250) — matrix operand order repaired and checked by reading, not compiled, not run on hardware
//
#include <hip/hip_runtime.h>
#include <hip/hip_bf16.h>

typedef _Float16 half8  __attribute__((ext_vector_type(8)));
typedef _Float16 half16 __attribute__((ext_vector_type(16)));
typedef float    f32x8  __attribute__((ext_vector_type(8)));
typedef float    f32x4  __attribute__((ext_vector_type(4)));

#define HDIM 256
#define LSTR 264   // padded LDS row stride (f16) for 256-wide activations (conflict-free)
#define SSTR 72    // padded LDS row stride (f16) for 64-wide layer-0 staging chunks

union H16 { half16 v; half8 h[2]; };

__device__ __forceinline__ f32x8 wmma16(half16 a, half16 b, f32x8 c) {
    return __builtin_amdgcn_wmma_f32_16x16x32_f16(false, a, false, b, (short)0, c,
                                                  false, false);
}

// A fragment (16x32 f16) from LDS row-major tile. Per CDNA5 layout:
// lane L holds row M=L%16; elements 0..7 = K (L/16)*8+0..7, 8..15 = 16+(L/16)*8+0..7
__device__ __forceinline__ half16 load_a_frag(const _Float16* act, int stride, int kOff) {
    const int lane = threadIdx.x & 31;
    const int row  = lane & 15;
    const int ksub = (lane >> 4) * 8;
    const _Float16* p = act + row * stride + kOff;
    H16 a;
    a.h[0] = *(const half8*)(p + ksub);
    a.h[1] = *(const half8*)(p + 16 + ksub);
    return a.v;
}

// B fragment (32x16 f16) from pre-packed global weights: frag = 512 f16, lane-major.
// Lane L holds col N=L%16; elements 0..7 = K (L/16)*8+0..7, 8..15 = 16+(L/16)*8+0..7 (same as A).
__device__ __forceinline__ half16 load_b_frag(const _Float16* __restrict__ wf, int frag) {
    const int lane = threadIdx.x & 31;
    const _Float16* p = wf + (size_t)frag * 512 + lane * 16;
    H16 b;
    b.h[0] = *(const half8*)(p);
    b.h[1] = *(const half8*)(p + 8);
    __builtin_prefetch(p + 512, 0, 0);   // global_prefetch next fragment
    return b.v;
}

// C/D fragment element j of lane L is (M = j + 8*(L/16), N = L%16)
__device__ __forceinline__ void store_act(f32x8 c, _Float16* actOut, int nt, bool relu) {
    const int lane  = threadIdx.x & 31;
    const int col   = lane & 15;
    const int rhalf = lane >> 4;
#pragma unroll
    for (int j = 0; j < 8; ++j) {
        float v = c[j];
        if (relu) v = fmaxf(v, 0.0f);
        actOut[(j + 8 * rhalf) * LSTR + nt * 16 + col] = (_Float16)v;
    }
}

// One 256->256 layer entirely from/to LDS activations
__device__ __forceinline__ void layer256(const _Float16* actIn, _Float16* actOut,
                                         const _Float16* __restrict__ wf,
                                         const float* __restrict__ bias, bool relu) {
    const int col = threadIdx.x & 15;
    for (int nt = 0; nt < 16; ++nt) {
        float bv = bias[nt * 16 + col];
        f32x8 c = {bv, bv, bv, bv, bv, bv, bv, bv};
#pragma unroll
        for (int kc = 0; kc < 8; ++kc) {
            half16 a = load_a_frag(actIn, LSTR, kc * 32);
            half16 b = load_b_frag(wf, nt * 8 + kc);
            c = wmma16(a, b, c);
        }
        store_act(c, actOut, nt, relu);
    }
}

// ---------------- Edge block: gather -> 4-layer MLP -> residual + scatter-add ---------
__global__ __launch_bounds__(32, 1) void edge_block_kernel(
    const float* __restrict__ node_attr, const float* __restrict__ edge_attr,
    const long long* __restrict__ edge_index,
    const _Float16* __restrict__ wf0, const float* __restrict__ b0,
    const _Float16* __restrict__ wf1, const float* __restrict__ b1,
    const _Float16* __restrict__ wf2, const float* __restrict__ b2,
    const _Float16* __restrict__ wf3, const float* __restrict__ b3,
    float* __restrict__ agg, float* __restrict__ edge_res_out, int nEdges) {
    __shared__ _Float16 stage[16 * SSTR];
    __shared__ _Float16 actA[16 * LSTR];
    __shared__ _Float16 actB[16 * LSTR];
    __shared__ int recvS[16];

    const int lane = threadIdx.x & 31;
    const int col = lane & 15, rhalf = lane >> 4;
    const int e0 = blockIdx.x * 16;

    if (lane < 16) {
        int e = min(e0 + lane, nEdges - 1);
        recvS[lane] = (int)edge_index[nEdges + e];
    }

    // layer 0: K = 768 streamed in 12 chunks of 64; all 16 N-tiles accumulate in VGPRs
    f32x8 acc[16];
#pragma unroll
    for (int nt = 0; nt < 16; ++nt) {
        float bv = b0[nt * 16 + col];
        f32x8 ci = {bv, bv, bv, bv, bv, bv, bv, bv};
        acc[nt] = ci;
    }

    const int srow = lane >> 1, shalf = lane & 1;      // 2 threads stage each row
    const int eMine = min(e0 + srow, nEdges - 1);
    const long long snd = edge_index[eMine];
    const long long rcv = edge_index[nEdges + eMine];

    for (int kc = 0; kc < 12; ++kc) {
        const int g = kc * 64 + shalf * 32;            // global feature base (32-aligned)
        const float* src;
        if (g < 256)      src = node_attr + (size_t)snd * HDIM + g;
        else if (g < 512) src = node_attr + (size_t)rcv * HDIM + (g - 256);
        else              src = edge_attr + (size_t)eMine * HDIM + (g - 512);
        __syncthreads();
#pragma unroll
        for (int j = 0; j < 8; ++j) {
            f32x4 v = *(const f32x4*)(src + j * 4);
            _Float16* d = &stage[srow * SSTR + shalf * 32 + j * 4];
            d[0] = (_Float16)v.x; d[1] = (_Float16)v.y;
            d[2] = (_Float16)v.z; d[3] = (_Float16)v.w;
        }
        __syncthreads();
#pragma unroll
        for (int sub = 0; sub < 2; ++sub) {
            half16 a = load_a_frag(stage, SSTR, sub * 32);
#pragma unroll
            for (int nt = 0; nt < 16; ++nt) {
                half16 b = load_b_frag(wf0, nt * 24 + kc * 2 + sub);
                acc[nt] = wmma16(a, b, acc[nt]);
            }
        }
    }
#pragma unroll
    for (int nt = 0; nt < 16; ++nt) store_act(acc[nt], actB, nt, true);
    __syncthreads();

    layer256(actB, actA, wf1, b1, true);
    __syncthreads();
    layer256(actA, actB, wf2, b2, true);
    __syncthreads();

    // layer 3 (no ReLU) fused with residual write + scatter-add
    for (int nt = 0; nt < 16; ++nt) {
        float bv = b3[nt * 16 + col];
        f32x8 c = {bv, bv, bv, bv, bv, bv, bv, bv};
#pragma unroll
        for (int kc = 0; kc < 8; ++kc) {
            half16 a = load_a_frag(actB, LSTR, kc * 32);
            half16 b = load_b_frag(wf3, nt * 8 + kc);
            c = wmma16(a, b, c);
        }
#pragma unroll
        for (int j = 0; j < 8; ++j) {
            const int m = j + 8 * rhalf;
            const int e = e0 + m;
            const int n = nt * 16 + col;
            if (e < nEdges) {
                const float v = c[j];
                const size_t off = (size_t)e * HDIM + n;
                edge_res_out[off] = edge_attr[off] + v;
                atomicAdd(&agg[(size_t)recvS[m] * HDIM + n], v);
            }
        }
    }
}

// ---------------- Node block: concat(node_attr, agg) -> 4-layer MLP -> residual -------
__global__ __launch_bounds__(32, 1) void node_block_kernel(
    const float* __restrict__ node_attr, const float* __restrict__ agg,
    const _Float16* __restrict__ wf0, const float* __restrict__ b0,
    const _Float16* __restrict__ wf1, const float* __restrict__ b1,
    const _Float16* __restrict__ wf2, const float* __restrict__ b2,
    const _Float16* __restrict__ wf3, const float* __restrict__ b3,
    float* __restrict__ x_out, int nNodes) {
    __shared__ _Float16 stage[16 * SSTR];
    __shared__ _Float16 actA[16 * LSTR];
    __shared__ _Float16 actB[16 * LSTR];

    const int lane = threadIdx.x & 31;
    const int col = lane & 15, rhalf = lane >> 4;
    const int v0 = blockIdx.x * 16;

    f32x8 acc[16];
#pragma unroll
    for (int nt = 0; nt < 16; ++nt) {
        float bv = b0[nt * 16 + col];
        f32x8 ci = {bv, bv, bv, bv, bv, bv, bv, bv};
        acc[nt] = ci;
    }

    const int srow = lane >> 1, shalf = lane & 1;
    const int nMine = min(v0 + srow, nNodes - 1);

    for (int kc = 0; kc < 8; ++kc) {                    // K = 512 in 8 chunks of 64
        const int g = kc * 64 + shalf * 32;
        const float* src = (g < 256)
            ? node_attr + (size_t)nMine * HDIM + g
            : agg + (size_t)nMine * HDIM + (g - 256);
        __syncthreads();
#pragma unroll
        for (int j = 0; j < 8; ++j) {
            f32x4 v = *(const f32x4*)(src + j * 4);
            _Float16* d = &stage[srow * SSTR + shalf * 32 + j * 4];
            d[0] = (_Float16)v.x; d[1] = (_Float16)v.y;
            d[2] = (_Float16)v.z; d[3] = (_Float16)v.w;
        }
        __syncthreads();
#pragma unroll
        for (int sub = 0; sub < 2; ++sub) {
            half16 a = load_a_frag(stage, SSTR, sub * 32);
#pragma unroll
            for (int nt = 0; nt < 16; ++nt) {
                half16 b = load_b_frag(wf0, nt * 16 + kc * 2 + sub);
                acc[nt] = wmma16(a, b, acc[nt]);
            }
        }
    }
#pragma unroll
    for (int nt = 0; nt < 16; ++nt) store_act(acc[nt], actB, nt, true);
    __syncthreads();

    layer256(actB, actA, wf1, b1, true);
    __syncthreads();
    layer256(actA, actB, wf2, b2, true);
    __syncthreads();

    for (int nt = 0; nt < 16; ++nt) {
        float bv = b3[nt * 16 + col];
        f32x8 c = {bv, bv, bv, bv, bv, bv, bv, bv};
#pragma unroll
        for (int kc = 0; kc < 8; ++kc) {
            half16 a = load_a_frag(actB, LSTR, kc * 32);
            half16 b = load_b_frag(wf3, nt * 8 + kc);
            c = wmma16(a, b, c);
        }
#pragma unroll
        for (int j = 0; j < 8; ++j) {
            const int m = j + 8 * rhalf;
            const int nd = v0 + m;
            const int n = nt * 16 + col;
            if (nd < nNodes) {
                const size_t off = (size_t)nd * HDIM + n;
                x_out[off] = node_attr[off] + c[j];
            }
        }
    }
}

// Convert fp32 weight [O=256, I] into f16 B-fragments (frag = nt*(I/32)+kc, 512 f16
// lane-major). dst[t] with t = frag*512 + lane*16 + e; element (N,K) per B layout.
__global__ void convert_w_kernel(const float* __restrict__ W, _Float16* __restrict__ dst,
                                 int I) {
    int t = blockIdx.x * blockDim.x + threadIdx.x;
    if (t >= 256 * I) return;
    const int frag = t >> 9, r = t & 511;
    const int lane = r >> 4, e = r & 15;
    const int KC = I >> 5;
    const int nt = frag / KC, kc = frag % KC;
    const int n = nt * 16 + (lane & 15);
    const int k = kc * 32 + (lane >> 4) * 8 + (e & 7) + (e >> 3) * 16;
    dst[t] = (_Float16)W[n * I + k];
}

__global__ void zero_f32_kernel(float* __restrict__ p, size_t n) {
    size_t i = (size_t)blockIdx.x * blockDim.x + threadIdx.x;
    const size_t stride = (size_t)gridDim.x * blockDim.x;
    for (; i < n; i += stride) p[i] = 0.0f;
}

extern "C" void kernel_launch(void* const* d_in, const int* in_sizes, int n_in,
                              void* d_out, int out_size, void* d_ws, size_t ws_size,
                              hipStream_t stream) {
    const float* node_attr = (const float*)d_in[0];
    const float* edge_attr = (const float*)d_in[1];
    const long long* edge_index = (const long long*)d_in[2];
    const float* ew[4] = {(const float*)d_in[3], (const float*)d_in[5],
                          (const float*)d_in[7], (const float*)d_in[9]};
    const float* eb[4] = {(const float*)d_in[4], (const float*)d_in[6],
                          (const float*)d_in[8], (const float*)d_in[10]};
    const float* nw[4] = {(const float*)d_in[11], (const float*)d_in[13],
                          (const float*)d_in[15], (const float*)d_in[17]};
    const float* nb[4] = {(const float*)d_in[12], (const float*)d_in[14],
                          (const float*)d_in[16], (const float*)d_in[18]};

    const int nNodes = in_sizes[0] / HDIM;
    const int nEdges = in_sizes[1] / HDIM;

    // workspace layout: f16 weight fragments, then f32 agg buffer
    char* ws = (char*)d_ws;
    size_t off = 0;
    const int eI[4] = {768, 256, 256, 256};
    const int nI[4] = {512, 256, 256, 256};
    _Float16* ewf[4];
    _Float16* nwf[4];
    for (int i = 0; i < 4; ++i) { ewf[i] = (_Float16*)(ws + off); off += (size_t)256 * eI[i] * 2; }
    for (int i = 0; i < 4; ++i) { nwf[i] = (_Float16*)(ws + off); off += (size_t)256 * nI[i] * 2; }
    float* agg = (float*)(ws + off);

    for (int i = 0; i < 4; ++i) {
        int te = 256 * eI[i];
        convert_w_kernel<<<(te + 255) / 256, 256, 0, stream>>>(ew[i], ewf[i], eI[i]);
        int tn = 256 * nI[i];
        convert_w_kernel<<<(tn + 255) / 256, 256, 0, stream>>>(nw[i], nwf[i], nI[i]);
    }

    zero_f32_kernel<<<4096, 256, 0, stream>>>(agg, (size_t)nNodes * HDIM);

    float* out_x = (float*)d_out;
    float* out_e = out_x + (size_t)nNodes * HDIM;

    edge_block_kernel<<<(nEdges + 15) / 16, 32, 0, stream>>>(
        node_attr, edge_attr, edge_index,
        ewf[0], eb[0], ewf[1], eb[1], ewf[2], eb[2], ewf[3], eb[3],
        agg, out_e, nEdges);

    node_block_kernel<<<(nNodes + 15) / 16, 32, 0, stream>>>(
        node_attr, agg,
        nwf[0], nb[0], nwf[1], nb[1], nwf[2], nb[2], nwf[3], nb[3],
        out_x, nNodes);
}
